// Model_83803401879727
// MI455X (gfx1250) — matrix operand order repaired and checked by reading, not compiled, not run on hardware
//
#include <hip/hip_runtime.h>

#define H 128
#define NS_N 20000
#define NE_N 100000
#define NT_N 200000
#define NEDGE 1000000
#define NLAB 500000

typedef __attribute__((ext_vector_type(16))) __bf16        v16bf;
typedef __attribute__((ext_vector_type(8)))  float         v8f;
typedef __attribute__((ext_vector_type(8)))  unsigned int  v8u;

// ---- bf16 helpers (round-to-nearest-even) ----
__device__ __forceinline__ unsigned short f2bf(float x) {
  unsigned u = __float_as_uint(x);
  unsigned r = (u + 0x7FFFu + ((u >> 16) & 1u)) >> 16;
  return (unsigned short)r;
}
__device__ __forceinline__ float bf2f(unsigned short h) {
  return __uint_as_float(((unsigned)h) << 16);
}

// =====================================================================
// Edge scatter: agg[dst] += x[src], cnt[dst] += 1.   One wave per edge,
// each lane handles 4 consecutive floats (32*4 = 128 = H).
// =====================================================================
__global__ __launch_bounds__(256)
void scatter_mean_accum(const float* __restrict__ X,
                        const int* __restrict__ src, const int* __restrict__ dst,
                        float* __restrict__ agg, float* __restrict__ cnt, int nEdges)
{
  int t = blockIdx.x * 256 + threadIdx.x;
  int e = t >> 5, lane = t & 31;
  if (e >= nEdges) return;
  int s = src[e], d = dst[e];
  const float4 xv = *(const float4*)(X + (size_t)s * H + lane * 4);
  float* dp = agg + (size_t)d * H + lane * 4;
  atomicAdd(dp + 0, xv.x);
  atomicAdd(dp + 1, xv.y);
  atomicAdd(dp + 2, xv.z);
  atomicAdd(dp + 3, xv.w);
  if (lane == 0) atomicAdd(cnt + d, 1.0f);
}

// =====================================================================
// Out[N x 128] (op)= (rowscale * A[N x 128]) @ W[128 x 128]^T  (+ bias)
// flags: bit0 = accumulate into Out, bit1 = ReLU at the end.
// cnt != nullptr -> rowscale = 1/max(cnt[row],1)  (segment mean).
// bf16 hi/lo compensated WMMA: A@B ~= AhBh + AhBl + AlBh (fp32-like).
// 256 threads = 8 waves; wave w handles output rows [strip*16, strip*16+16).
// B fragments (hi & lo) for all 8 N-tiles x 4 K-chunks are pre-packed
// into LDS in the exact wave32 WMMA VGPR layout.
// =====================================================================
__global__ __launch_bounds__(256)
void gemm128(const float* __restrict__ A, const float* __restrict__ cnt,
             const float* __restrict__ W, const float* __restrict__ bias,
             float* __restrict__ Out, int nRows, int flags)
{
  __shared__ unsigned int bfr[64 * 256];   // 64 KB: 32 (tile,chunk) x {hi,lo} x 256 dwords

  const int tid = threadIdx.x;

  // ---- cooperatively build B fragments: B[k][n] = W[n][k] ----
  for (int idx = tid; idx < 8192; idx += 256) {
    int e = idx & 255;        // lane-major fragment slot: e = v*32 + lane
    int g = idx >> 8;         // g = tile*4 + chunk
    int v  = e >> 5;
    int ln = e & 31;
    int half = ln >> 4;
    int n = (g >> 2) * 16 + (ln & 15);
    int k = (g & 3) * 32 + ((v < 4) ? 2 * v : 16 + 2 * (v - 4)) + 8 * half;
    float w0 = W[(size_t)n * H + k];
    float w1 = W[(size_t)n * H + k + 1];
    unsigned short h0 = f2bf(w0), h1 = f2bf(w1);
    unsigned short l0 = f2bf(w0 - bf2f(h0)), l1 = f2bf(w1 - bf2f(h1));
    bfr[(g * 2 + 0) * 256 + e] = ((unsigned)h1 << 16) | h0;
    bfr[(g * 2 + 1) * 256 + e] = ((unsigned)l1 << 16) | l0;
  }
  __syncthreads();

  const int wave = tid >> 5;
  const int lane = tid & 31;
  const int strip = blockIdx.x * 8 + wave;
  if (strip * 16 >= nRows) return;          // wave-uniform: EXEC stays all-ones

  const int half = lane >> 4;
  const int mrow = strip * 16 + (lane & 15);

  float rs = 1.0f;
  if (cnt) rs = 1.0f / fmaxf(cnt[mrow], 1.0f);

  // ---- A fragments (hi/lo) for the 4 K-chunks, ISA 16-bit A layout ----
  v8u ahP[4], alP[4];
  const float* Arow = A + (size_t)mrow * H;
#pragma unroll
  for (int c = 0; c < 4; ++c) {
#pragma unroll
    for (int v = 0; v < 8; ++v) {
      int k = c * 32 + ((v < 4) ? 2 * v : 16 + 2 * (v - 4)) + 8 * half;
      float x0 = Arow[k] * rs;
      float x1 = Arow[k + 1] * rs;
      unsigned short h0 = f2bf(x0), h1 = f2bf(x1);
      unsigned short l0 = f2bf(x0 - bf2f(h0)), l1 = f2bf(x1 - bf2f(h1));
      ahP[c][v] = ((unsigned)h1 << 16) | h0;
      alP[c][v] = ((unsigned)l1 << 16) | l0;
    }
  }

#pragma unroll 1
  for (int tile = 0; tile < 8; ++tile) {
    v8f acc = {};
#pragma unroll
    for (int c = 0; c < 4; ++c) {
      int g = tile * 4 + c;
      v8u bh, bl;
#pragma unroll
      for (int v = 0; v < 8; ++v) {
        bh[v] = bfr[(g * 2 + 0) * 256 + v * 32 + lane];
        bl[v] = bfr[(g * 2 + 1) * 256 + v * 32 + lane];
      }
      v16bf ah = __builtin_bit_cast(v16bf, ahP[c]);
      v16bf al = __builtin_bit_cast(v16bf, alP[c]);
      v16bf Bh = __builtin_bit_cast(v16bf, bh);
      v16bf Bl = __builtin_bit_cast(v16bf, bl);
      acc = __builtin_amdgcn_wmma_f32_16x16x32_bf16(false, ah, false, Bh, (short)0, acc, false, false);
      acc = __builtin_amdgcn_wmma_f32_16x16x32_bf16(false, ah, false, Bl, (short)0, acc, false, false);
      acc = __builtin_amdgcn_wmma_f32_16x16x32_bf16(false, al, false, Bh, (short)0, acc, false, false);
    }
    // ---- C/D layout store: lanes 0-15 -> M=r, lanes 16-31 -> M=r+8 ----
    int ncol = tile * 16 + (lane & 15);
    float badd = bias ? bias[ncol] : 0.0f;
#pragma unroll
    for (int r = 0; r < 8; ++r) {
      int m = strip * 16 + r + 8 * half;
      float val = acc[r] + badd;
      size_t off = (size_t)m * H + ncol;
      if (flags & 1) val += Out[off];
      if (flags & 2) val = fmaxf(val, 0.0f);
      Out[off] = val;
    }
  }
}

// =====================================================================
// Combine the two team-targeting relations' self weights/biases.
// =====================================================================
__global__ __launch_bounds__(256)
void combine_wb(const float* __restrict__ Wa, const float* __restrict__ Wb,
                const float* __restrict__ ba, const float* __restrict__ bb,
                float* __restrict__ Wc, float* __restrict__ bc)
{
  int i = blockIdx.x * 256 + threadIdx.x;
  if (i < H * H) Wc[i] = Wa[i] + Wb[i];
  if (i < H)     bc[i] = ba[i] + bb[i];
}

// =====================================================================
// Classifier: pred[e] = dot(He[ie[e]], Ht[it[e]]).  One wave per edge.
// =====================================================================
__global__ __launch_bounds__(256)
void dot_classifier(const float* __restrict__ He, const float* __restrict__ Ht,
                    const int* __restrict__ ie, const int* __restrict__ it,
                    float* __restrict__ out, int nL)
{
  int t = blockIdx.x * 256 + threadIdx.x;
  int e = t >> 5, lane = t & 31;
  if (e >= nL) return;
  const float4 a = *(const float4*)(He + (size_t)ie[e] * H + lane * 4);
  const float4 b = *(const float4*)(Ht + (size_t)it[e] * H + lane * 4);
  float s = a.x * b.x + a.y * b.y + a.z * b.z + a.w * b.w;
#pragma unroll
  for (int off = 16; off; off >>= 1) s += __shfl_xor(s, off, 32);
  if (lane == 0) out[e] = s;
}

// =====================================================================
extern "C" void kernel_launch(void* const* d_in, const int* in_sizes, int n_in,
                              void* d_out, int out_size, void* d_ws, size_t ws_size,
                              hipStream_t stream)
{
  const float* emb_s = (const float*)d_in[0];
  const float* emb_e = (const float*)d_in[1];
  const float* emb_t = (const float*)d_in[2];
  const float* Wl1 = (const float*)d_in[3];
  const float* Wr1 = (const float*)d_in[4];
  const float* b1  = (const float*)d_in[5];
  const float* Wl2 = (const float*)d_in[6];
  const float* Wr2 = (const float*)d_in[7];
  const float* b2  = (const float*)d_in[8];
  const int* te_s = (const int*)d_in[9];   const int* te_d = (const int*)d_in[10];
  const int* et_s = (const int*)d_in[11];  const int* et_d = (const int*)d_in[12];
  const int* st_s = (const int*)d_in[13];  const int* st_d = (const int*)d_in[14];
  const int* ts_s = (const int*)d_in[15];  const int* ts_d = (const int*)d_in[16];
  const int* eli_t = (const int*)d_in[17];
  const int* eli_e = (const int*)d_in[18];
  float* pred = (float*)d_out;

  // ---- workspace carve (~421 MB of f32) ----
  float* p    = (float*)d_ws;
  float* agg  = p; p += (size_t)NT_N * H;   // reused by every relation pass
  float* cntb = p; p += NT_N;
  float* h1s  = p; p += (size_t)NS_N * H;
  float* h1e  = p; p += (size_t)NE_N * H;
  float* h1t  = p; p += (size_t)NT_N * H;
  float* h2e  = p; p += (size_t)NE_N * H;
  float* h2t  = p; p += (size_t)NT_N * H;
  float* Wc   = p; p += H * H;
  float* bc   = p; p += H;

  const size_t WSTEP = (size_t)H * H;   // per-relation weight slice
  const int ACC = 1, RELU = 2;

  auto gemm = [&](const float* A, const float* c, const float* W, const float* b,
                  float* O, int N, int flags) {
    int blocks = (N / 16 + 7) / 8;
    gemm128<<<blocks, 256, 0, stream>>>(A, c, W, b, O, N, flags);
  };
  auto relpass = [&](const float* X, const int* si, const int* di, int Nd,
                     const float* Wl, float* O, int flags) {
    hipMemsetAsync(agg, 0, (size_t)Nd * H * sizeof(float), stream);
    hipMemsetAsync(cntb, 0, (size_t)Nd * sizeof(float), stream);
    scatter_mean_accum<<<(NEDGE * 32) / 256, 256, 0, stream>>>(X, si, di, agg, cntb, NEDGE);
    gemm(agg, cntb, Wl, nullptr, O, Nd, flags);
  };

  // ================= Layer 1 =================
  combine_wb<<<(H * H + 255) / 256, 256, 0, stream>>>(Wr1 + 1 * WSTEP, Wr1 + 2 * WSTEP,
                                                      b1 + 1 * H, b1 + 2 * H, Wc, bc);
  // self terms: out = x_dst @ Wr^T + b
  gemm(emb_e, nullptr, Wr1 + 0 * WSTEP, b1 + 0 * H, h1e, NE_N, 0);
  gemm(emb_t, nullptr, Wc,              bc,         h1t, NT_N, 0);
  gemm(emb_s, nullptr, Wr1 + 3 * WSTEP, b1 + 3 * H, h1s, NS_N, 0);
  // relation terms: out += mean_agg @ Wl^T  (ReLU fused on last pass per type)
  relpass(emb_t, te_s, te_d, NE_N, Wl1 + 0 * WSTEP, h1e, ACC | RELU);  // team->expert
  relpass(emb_e, et_s, et_d, NT_N, Wl1 + 1 * WSTEP, h1t, ACC);         // expert->team
  relpass(emb_s, st_s, st_d, NT_N, Wl1 + 2 * WSTEP, h1t, ACC | RELU);  // skill->team
  relpass(emb_t, ts_s, ts_d, NS_N, Wl1 + 3 * WSTEP, h1s, ACC | RELU);  // team->skill

  // ================= Layer 2 (skill output unused by classifier -> skipped) =================
  combine_wb<<<(H * H + 255) / 256, 256, 0, stream>>>(Wr2 + 1 * WSTEP, Wr2 + 2 * WSTEP,
                                                      b2 + 1 * H, b2 + 2 * H, Wc, bc);
  gemm(h1e, nullptr, Wr2 + 0 * WSTEP, b2 + 0 * H, h2e, NE_N, 0);
  gemm(h1t, nullptr, Wc,              bc,         h2t, NT_N, 0);
  relpass(h1t, te_s, te_d, NE_N, Wl2 + 0 * WSTEP, h2e, ACC);   // team->expert
  relpass(h1e, et_s, et_d, NT_N, Wl2 + 1 * WSTEP, h2t, ACC);   // expert->team
  relpass(h1s, st_s, st_d, NT_N, Wl2 + 2 * WSTEP, h2t, ACC);   // skill->team

  // ================= Classifier =================
  dot_classifier<<<(NLAB * 32) / 256, 256, 0, stream>>>(h2e, h2t, eli_e, eli_t, pred, NLAB);
}
